// MultiHeadAttentionBlock_24927990186425
// MI455X (gfx1250) — matrix operand order repaired and checked by reading, compiled, not run on hardware
//
#include <hip/hip_runtime.h>
#include <hip/hip_bf16.h>

// MHA block: B=2, L=2048, D=512, H=16, DK=32.
// MI455X strategy: all live data (45MB) is L2-resident (192MB), so GEMMs read
// f16 operands straight from global with b128 loads in WMMA fragment layout
// (no LDS staging). Attention is flash-style (online softmax) so the 512MB
// logits tensor never exists. Softmax is done row-per-lane (A-fragment chunk
// order) so P fragments are built in registers and cross-lane traffic is
// 1 xor-16 shuffle per reduction instead of 4-deep trees per row.

#define B_  2
#define L_  2048
#define D_  512
#define H_  16
#define DK_ 32

typedef __attribute__((ext_vector_type(16))) _Float16 v16h;
typedef __attribute__((ext_vector_type(8)))  _Float16 v8h;
typedef __attribute__((ext_vector_type(4)))  _Float16 v4h;
typedef __attribute__((ext_vector_type(8)))  float    v8f;

union AF { v16h v; v8h c[2]; };

static __device__ __forceinline__ v8f wmma16(const AF& a, const AF& b, v8f c) {
  return __builtin_amdgcn_wmma_f32_16x16x32_f16(false, a.v, false, b.v,
                                                (short)0, c, false, false);
}

// ---------------------------------------------------------------------------
// Pre-pass 1: elementwise f32 -> f16 (coalesced b128 in, b128 out).
// ---------------------------------------------------------------------------
__global__ void cvt_f16_kernel(const float* __restrict__ src,
                               _Float16* __restrict__ dst) {
  const size_t i = ((size_t)blockIdx.x * 256 + threadIdx.x) * 8;
  float4 a = *(const float4*)(src + i);
  float4 b = *(const float4*)(src + i + 4);
  v8h o;
  o[0] = (_Float16)a.x; o[1] = (_Float16)a.y;
  o[2] = (_Float16)a.z; o[3] = (_Float16)a.w;
  o[4] = (_Float16)b.x; o[5] = (_Float16)b.y;
  o[6] = (_Float16)b.z; o[7] = (_Float16)b.w;
  *(v8h*)(dst + i) = o;
}

// ---------------------------------------------------------------------------
// Pre-pass 2: W[k][n] f32 -> Wt[n][k] f16 via 32x32 LDS tile (done once, so
// every GEMM B-fragment is two contiguous 16B runs of k-values).
// ---------------------------------------------------------------------------
__global__ void wt_kernel(const float* __restrict__ W,
                          _Float16* __restrict__ Wt) {
  __shared__ float T[32][33];
  const int t  = threadIdx.x;            // 256 threads
  const int k0 = blockIdx.x * 32;
  const int n0 = blockIdx.y * 32;
  {
    const int kr = t >> 3;
    const int nc = (t & 7) * 4;
    float4 a = *(const float4*)(W + (size_t)(k0 + kr) * D_ + n0 + nc);
    T[kr][nc + 0] = a.x; T[kr][nc + 1] = a.y;
    T[kr][nc + 2] = a.z; T[kr][nc + 3] = a.w;
  }
  __syncthreads();
  {
    const int nr = t >> 3;
    const int kc = (t & 7) * 4;
    v4h o;
    o[0] = (_Float16)T[kc + 0][nr];
    o[1] = (_Float16)T[kc + 1][nr];
    o[2] = (_Float16)T[kc + 2][nr];
    o[3] = (_Float16)T[kc + 3][nr];
    *(v4h*)(Wt + (size_t)(n0 + nr) * D_ + k0 + kc) = o;
  }
}

// ---------------------------------------------------------------------------
// GEMM: C = A(f16,[M][512]) @ Wt(f16,[n][k]) + bias.  No LDS, no barriers:
// per wave, one 16x64 strip; per K-step: 2 A b128 + 8 B b128 + 4 WMMA.
// MODE 0: dst f16 [((b*H+h)*L + l)*DK + d]   (Q, K head-major)
// MODE 2: dst f16 [((b*H+h)*DK + d)*L + l]   (V transposed)
// MODE 3: dst f32 [m*D + n]                  (final output)
// ---------------------------------------------------------------------------
template<int MODE>
__global__ void gemm_kernel(const _Float16* __restrict__ A,
                            const _Float16* __restrict__ Wt,
                            const float* __restrict__ bias,
                            void* __restrict__ dstv) {
  const int wave = threadIdx.x >> 5;
  const int lane = threadIdx.x & 31;
  const int hi   = lane >> 4;
  const int lm   = lane & 15;
  const int mt   = blockIdx.x * 8 + wave;   // 16-row tile index
  const int n0   = blockIdx.y * 64;

  const _Float16* arow = A + (size_t)(mt * 16 + lm) * D_;

  v8f acc[4] = {{}, {}, {}, {}};

  for (int ks = 0; ks < D_ / 32; ++ks) {
    const int k0 = ks * 32;
    AF af;
    af.c[0] = *(const v8h*)(arow + k0 + 8 * hi);
    af.c[1] = *(const v8h*)(arow + k0 + 16 + 8 * hi);
#pragma unroll
    for (int t = 0; t < 4; ++t) {
      const _Float16* brow = Wt + (size_t)(n0 + t * 16 + lm) * D_ + k0;
      AF bf;
      bf.c[0] = *(const v8h*)(brow + 8 * hi);
      bf.c[1] = *(const v8h*)(brow + 16 + 8 * hi);
      acc[t] = wmma16(af, bf, acc[t]);
    }
  }

#pragma unroll
  for (int t = 0; t < 4; ++t) {
#pragma unroll
    for (int r = 0; r < 8; ++r) {
      const int m = mt * 16 + r + 8 * hi;   // row over B*L
      const int n = n0 + t * 16 + lm;       // col over D
      const float val = acc[t][r] + bias[n];
      if constexpr (MODE == 3) {
        ((float*)dstv)[(size_t)m * D_ + n] = val;
      } else {
        _Float16* dst = (_Float16*)dstv;
        const int b = m >> 11;       // / L_
        const int l = m & (L_ - 1);
        const int h = n >> 5;        // / DK_
        const int d = n & (DK_ - 1);
        if constexpr (MODE == 2)
          dst[(((size_t)b * H_ + h) * DK_ + d) * L_ + l] = (_Float16)val;
        else
          dst[(((size_t)b * H_ + h) * L_ + l) * DK_ + d] = (_Float16)val;
      }
    }
  }
}

// ---------------------------------------------------------------------------
// Flash attention: one head per blockIdx.y, 64 query rows per block,
// 4 waves x 16 rows, streaming keys in steps of 32 with online softmax.
// Row softmax runs row-per-lane: lane (lm,hi) owns row q0+lm, k-chunks
// [8hi,8hi+8) u [16+8hi,+8) -- exactly the WMMA A-fragment order, so the
// exp'd P fragment is assembled directly in registers.
// ---------------------------------------------------------------------------
__global__ void flash_attn_kernel(const _Float16* __restrict__ Qh,
                                  const _Float16* __restrict__ Kh,
                                  const _Float16* __restrict__ Vt,
                                  const unsigned char* __restrict__ mask,
                                  _Float16* __restrict__ Oi) {
  __shared__ __align__(16) float Sf[4][16][32];   // per-wave raw scores
  const int wave = threadIdx.x >> 5;
  const int lane = threadIdx.x & 31;
  const int hi   = lane >> 4;
  const int lm   = lane & 15;
  const int bh   = blockIdx.y;       // b*H + h
  const int b    = bh >> 4;
  const int h    = bh & 15;
  const int q0   = blockIdx.x * 64 + wave * 16;

  const _Float16* Qbase = Qh + (size_t)bh * L_ * DK_;
  const _Float16* Kbase = Kh + (size_t)bh * L_ * DK_;
  const _Float16* Vbase = Vt + (size_t)bh * DK_ * L_;
  // This lane's query-row mask (contiguous over keys).
  const unsigned char* mrow =
      mask + (size_t)b * L_ * L_ + (size_t)(q0 + lm) * L_;

  // Q fragment (A layout): row q0+lm, d-chunks [8*hi,+8) and [16+8*hi,+8)
  AF qf;
  {
    const _Float16* qrow = Qbase + (size_t)(q0 + lm) * DK_;
    qf.c[0] = *(const v8h*)(qrow + 8 * hi);
    qf.c[1] = *(const v8h*)(qrow + 16 + 8 * hi);
  }

  v8f acc0 = {}, acc1 = {};         // O accumulators (C-layout), d-halves
  float mrun = -1e30f, lrun = 0.0f; // per-lane: row q0+lm running max / sum

  const float scale = 0.17677669529663689f;  // 1/sqrt(DK)

  for (int kt = 0; kt < L_ / 32; ++kt) {
    const int key0 = kt * 32;

    if (kt + 1 < L_ / 32)
      __builtin_prefetch(Kbase + (size_t)(key0 + 32 + lm) * DK_, 0, 1);

    // Two 16-key score tiles: S = Q (16x32) x K^T, K-dim = DK = 32
    v8f s0 = {}, s1 = {};
    {
      AF kf;
      const _Float16* kr0 = Kbase + (size_t)(key0 + lm) * DK_;
      kf.c[0] = *(const v8h*)(kr0 + 8 * hi);
      kf.c[1] = *(const v8h*)(kr0 + 16 + 8 * hi);
      s0 = wmma16(qf, kf, s0);
      const _Float16* kr1 = Kbase + (size_t)(key0 + 16 + lm) * DK_;
      kf.c[0] = *(const v8h*)(kr1 + 8 * hi);
      kf.c[1] = *(const v8h*)(kr1 + 16 + 8 * hi);
      s1 = wmma16(qf, kf, s1);
    }

    // C-layout -> row-per-lane via f32 LDS round trip.
#pragma unroll
    for (int r = 0; r < 8; ++r) {
      Sf[wave][r + 8 * hi][lm]      = s0[r];
      Sf[wave][r + 8 * hi][16 + lm] = s1[r];
    }
    asm volatile("s_wait_dscnt 0x0" ::: "memory");

    const float* srow = &Sf[wave][lm][0];
    float4 x0 = *(const float4*)(srow + 8 * hi);
    float4 x1 = *(const float4*)(srow + 8 * hi + 4);
    float4 x2 = *(const float4*)(srow + 16 + 8 * hi);
    float4 x3 = *(const float4*)(srow + 16 + 8 * hi + 4);
    float vv[16] = {x0.x, x0.y, x0.z, x0.w, x1.x, x1.y, x1.z, x1.w,
                    x2.x, x2.y, x2.z, x2.w, x3.x, x3.y, x3.z, x3.w};

    // Mask: 16 bytes for this row's k-chunks as two 8-byte loads.
    uint2 ma = *(const uint2*)(mrow + key0 + 8 * hi);
    uint2 mb = *(const uint2*)(mrow + key0 + 16 + 8 * hi);
    const unsigned mw[4] = {ma.x, ma.y, mb.x, mb.y};
#pragma unroll
    for (int i = 0; i < 16; ++i) {
      const unsigned byte = (mw[i >> 2] >> ((i & 3) * 8)) & 0xffu;
      vv[i] = byte ? vv[i] * scale : -1e9f;
    }

    // Row max: local 16-tree + one xor-16 exchange with the other half-row.
    float m16 = vv[0];
#pragma unroll
    for (int i = 1; i < 16; ++i) m16 = fmaxf(m16, vv[i]);
    m16 = fmaxf(m16, __shfl_xor(m16, 16));
    const float mnew = fmaxf(mrun, m16);

    // exp + row sum; P fragment built directly in A-fragment order.
    AF pf;
    float ps = 0.0f;
#pragma unroll
    for (int i = 0; i < 16; ++i) {
      const float p = __expf(vv[i] - mnew);
      ps += p;
      pf.v[i] = (_Float16)p;
    }
    ps += __shfl_xor(ps, 16);

    const float corr = __expf(mrun - mnew);
    lrun = lrun * corr + ps;
    mrun = mnew;

    // Broadcast correction to C-layout accumulator rows (8 shuffles).
#pragma unroll
    for (int r = 0; r < 8; ++r) {
      const float c = __shfl(corr, r + 8 * hi);
      acc0[r] *= c;
      acc1[r] *= c;
    }

    // O += P (16x32 keys) x V; V-frag col n = d, k = keys (rows of Vt).
    AF vf;
    const _Float16* vr0 = Vbase + (size_t)lm * L_ + key0;
    vf.c[0] = *(const v8h*)(vr0 + 8 * hi);
    vf.c[1] = *(const v8h*)(vr0 + 16 + 8 * hi);
    acc0 = wmma16(pf, vf, acc0);
    const _Float16* vr1 = Vbase + (size_t)(16 + lm) * L_ + key0;
    vf.c[0] = *(const v8h*)(vr1 + 8 * hi);
    vf.c[1] = *(const v8h*)(vr1 + 16 + 8 * hi);
    acc1 = wmma16(pf, vf, acc1);
  }

  // Normalize (broadcast 1/l back to C-layout) and emit f16 in [B, L, D].
  const float linv = 1.0f / lrun;
#pragma unroll
  for (int r = 0; r < 8; ++r) {
    const float inv = __shfl(linv, r + 8 * hi);
    const int qrow = q0 + r + 8 * hi;
    const size_t o = ((size_t)b * L_ + qrow) * D_ + h * DK_;
    Oi[o + lm]      = (_Float16)(acc0[r] * inv);
    Oi[o + 16 + lm] = (_Float16)(acc1[r] * inv);
  }
}

// ---------------------------------------------------------------------------
extern "C" void kernel_launch(void* const* d_in, const int* in_sizes, int n_in,
                              void* d_out, int out_size, void* d_ws, size_t ws_size,
                              hipStream_t stream) {
  const float* q  = (const float*)d_in[0];
  const float* k  = (const float*)d_in[1];
  const float* v  = (const float*)d_in[2];
  const unsigned char* mask = (const unsigned char*)d_in[3];
  // d_in[4] = is_training (ignored; inference path)
  const float* wq = (const float*)d_in[5];
  const float* bq = (const float*)d_in[6];
  const float* wk = (const float*)d_in[7];
  const float* bk = (const float*)d_in[8];
  const float* wv = (const float*)d_in[9];
  const float* bv = (const float*)d_in[10];
  const float* wo = (const float*)d_in[11];
  const float* bo = (const float*)d_in[12];
  float* out = (float*)d_out;

  const size_t tE = (size_t)B_ * L_ * D_;     // 2M elems (per activation)
  const size_t wE = (size_t)D_ * D_;          // 256K elems (per weight)
  _Float16* Qh  = (_Float16*)d_ws;            // [B,H,L,DK]
  _Float16* Kh  = Qh  + tE;                   // [B,H,L,DK]
  _Float16* Vt  = Kh  + tE;                   // [B,H,DK,L]
  _Float16* Oi  = Vt  + tE;                   // [B,L,D]
  _Float16* X16q = Oi  + tE;
  _Float16* X16k = X16q + tE;
  _Float16* X16v = X16k + tE;
  _Float16* Wtq = X16v + tE;                  // [n][k] f16
  _Float16* Wtk = Wtq + wE;
  _Float16* Wtv = Wtk + wE;
  _Float16* Wto = Wtv + wE;

  // Pre-pass: f16 conversions (coalesced) + weight transposes (once).
  const int cvtBlocks = (int)(tE / 2048);     // 8 elems/thread, 256 thr
  cvt_f16_kernel<<<cvtBlocks, 256, 0, stream>>>(q, X16q);
  cvt_f16_kernel<<<cvtBlocks, 256, 0, stream>>>(k, X16k);
  cvt_f16_kernel<<<cvtBlocks, 256, 0, stream>>>(v, X16v);
  dim3 gwt(D_ / 32, D_ / 32);
  wt_kernel<<<gwt, 256, 0, stream>>>(wq, Wtq);
  wt_kernel<<<gwt, 256, 0, stream>>>(wk, Wtk);
  wt_kernel<<<gwt, 256, 0, stream>>>(wv, Wtv);
  wt_kernel<<<gwt, 256, 0, stream>>>(wo, Wto);

  // Projections (pure global-load WMMA GEMMs).
  dim3 gg((size_t)B_ * L_ / 128, D_ / 64);    // 32 x 8, 256 thr (8 waves)
  gemm_kernel<0><<<gg, 256, 0, stream>>>(X16q, Wtq, bq, (void*)Qh);
  gemm_kernel<0><<<gg, 256, 0, stream>>>(X16k, Wtk, bk, (void*)Kh);
  gemm_kernel<2><<<gg, 256, 0, stream>>>(X16v, Wtv, bv, (void*)Vt);

  // Flash attention.
  dim3 gattn(L_ / 64, B_ * H_);               // 32 x 32, 128 thr (4 waves)
  flash_attn_kernel<<<gattn, 128, 0, stream>>>(Qh, Kh, Vt, mask, Oi);

  // Output projection to fp32.
  gemm_kernel<3><<<gg, 256, 0, stream>>>(Oi, Wto, bo, (void*)out);
}
